// RaftModule_76914274337028
// MI455X (gfx1250) — compile-verified, matrix operand order checked by reading, not run on hardware
//
#include <hip/hip_runtime.h>
#include <math.h>

// ---------------------------------------------------------------------------
// MI455X (gfx1250) RAFT forward.
// Heavy conv/GEMM -> V_WMMA_F32_16X16X32_F16, 4 waves/block, 2 accum/wave,
// b128 LDS fragment reads, async global->LDS staging for the corr GEMM.
// ---------------------------------------------------------------------------

typedef __attribute__((ext_vector_type(16))) _Float16 v16h;
typedef __attribute__((ext_vector_type(8)))  _Float16 v8h;
typedef __attribute__((ext_vector_type(8)))  float    v8f;
typedef __attribute__((ext_vector_type(4)))  int      i32x4;

#if defined(__gfx1250__) && __has_builtin(__builtin_amdgcn_global_load_async_to_lds_b128) && __has_builtin(__builtin_amdgcn_s_wait_asynccnt)
#define RAFT_ASYNC_LDS 1
#else
#define RAFT_ASYNC_LDS 0
#endif

#if RAFT_ASYNC_LDS
__device__ __forceinline__ void async_cp16(const _Float16* g, _Float16* l)
{
    __builtin_amdgcn_global_load_async_to_lds_b128(
        (__attribute__((address_space(1))) i32x4*)g,
        (__attribute__((address_space(3))) i32x4*)l,
        0, 0);
}
#endif

// Build a 16-half WMMA fragment from one LDS row with two b128 reads.
// CDNA5 16-bit 16x32 layout: lane(lo,hi) holds halfs [8*hi..8*hi+7] and
// [16+8*hi..16+8*hi+7] of its row.
__device__ __forceinline__ v16h ld_frag(const _Float16* rowp, int hi)
{
    v8h alo = *(const v8h*)(rowp + 8 * hi);
    v8h ahi = *(const v8h*)(rowp + 16 + 8 * hi);
    v16h r;
#pragma unroll
    for (int i = 0; i < 8; ++i) { r[i] = alo[i]; r[i + 8] = ahi[i]; }
    return r;
}

// ===========================================================================
// Implicit-GEMM convolution with WMMA.
// Block = 128 threads (4 waves). Block tile: M=32 (spatial) x N=64 (Cout).
// Wave (mi = wid&1, nj = wid>>1) computes a 16x32 C tile: one A fragment,
// two B fragments, two WMMAs per K-step.
// fuse: 0=bias, 1=+relu, 2=+sigmoid, 3=+tanh, 4=0.25*(x+bias)
// ===========================================================================
__global__ __launch_bounds__(128)
void k_conv_wmma(const float* __restrict__ in, const float* __restrict__ wt,
                 const float* __restrict__ bias, float* __restrict__ out,
                 int N, int Cin, int H, int W, int Cout, int KH, int KW,
                 int stride, int padH, int padW, int OH, int OW, int fuse)
{
    const int tid   = threadIdx.x;
    const int lane  = tid & 31;
    const int wid   = tid >> 5;
    const int mi    = wid & 1;   // 16-row M subtile within block
    const int nj    = wid >> 1;  // 32-col N subtile within block
    const int tileM = blockIdx.x;
    const int tileN = blockIdx.y;
    const int n     = blockIdx.z;
    const int M     = OH * OW;
    const int KHW   = KH * KW;
    const int Ktot  = Cin * KHW;

    __shared__ alignas(16) _Float16 As[32][40];
    __shared__ alignas(16) _Float16 Bs[64][40];

    v8f acc0 = {};
    v8f acc1 = {};
    const int hi = lane >> 4;
    const int lo = lane & 15;

    for (int k0 = 0; k0 < Ktot; k0 += 32) {
        // prefetch next weight K-slab (weights are co-major contiguous Ktot)
        if (k0 + 32 < Ktot && tid < 64) {
            int co = tileN * 64 + tid;
            if (co < Cout)
                __builtin_prefetch(&wt[(size_t)co * Ktot + k0 + 32], 0, 1);
        }
        // stage A (im2col gather): 32 rows x 32 k, 8 elements per thread
        for (int t = tid; t < 1024; t += 128) {
            int row = t >> 5;
            int kk  = t & 31;
            int kg  = k0 + kk;
            float av = 0.f;
            if (kg < Ktot) {
                int ci = kg / KHW;
                int r  = kg - ci * KHW;
                int ky = r / KW;
                int kx = r - ky * KW;
                int m  = tileM * 32 + row;
                if (m < M) {
                    int oh = m / OW, ow = m - oh * OW;
                    int ih = oh * stride - padH + ky;
                    int iw = ow * stride - padW + kx;
                    if (ih >= 0 && ih < H && iw >= 0 && iw < W)
                        av = in[(((size_t)n * Cin + ci) * H + ih) * W + iw];
                }
            }
            As[row][kk] = (_Float16)av;
        }
        // stage B (weights): 64 rows x 32 k, 16 elements per thread
        for (int t = tid; t < 2048; t += 128) {
            int row = t >> 5;
            int kk  = t & 31;
            int kg  = k0 + kk;
            float bv = 0.f;
            int co = tileN * 64 + row;
            if (kg < Ktot && co < Cout)
                bv = wt[(size_t)co * Ktot + kg];   // ((co*Cin+ci)*KH+ky)*KW+kx
            Bs[row][kk] = (_Float16)bv;
        }
        __syncthreads();

        v16h a  = ld_frag(&As[mi * 16 + lo][0], hi);
        v16h b0 = ld_frag(&Bs[nj * 32 + lo][0], hi);
        v16h b1 = ld_frag(&Bs[nj * 32 + 16 + lo][0], hi);
        acc0 = __builtin_amdgcn_wmma_f32_16x16x32_f16(false, a, false, b0,
                                                      (short)0, acc0, false, false);
        acc1 = __builtin_amdgcn_wmma_f32_16x16x32_f16(false, a, false, b1,
                                                      (short)0, acc1, false, false);
        __syncthreads();
    }

    // C/D layout: lanes 0-15 -> N=lo, M=r ; lanes 16-31 -> N=lo, M=r+8.
    const int m_base = tileM * 32 + mi * 16;
#pragma unroll
    for (int half = 0; half < 2; ++half) {
        int co = tileN * 64 + nj * 32 + half * 16 + lo;
        if (co < Cout) {
            float bval = bias ? bias[co] : 0.f;
#pragma unroll
            for (int r = 0; r < 8; ++r) {
                float v = (half ? acc1[r] : acc0[r]) + bval;
                int m = m_base + r + 8 * hi;
                if (m < M) {
                    if (fuse == 1)      v = fmaxf(v, 0.f);
                    else if (fuse == 2) v = 1.f / (1.f + expf(-v));
                    else if (fuse == 3) v = tanhf(v);
                    else if (fuse == 4) v = 0.25f * v;
                    int oh = m / OW, ow = m - oh * OW;
                    out[(((size_t)n * Cout + co) * OH + oh) * OW + ow] = v;
                }
            }
        }
    }
}

// ===========================================================================
// Correlation GEMM, f16 inputs: C[m][n] = sum_k A[m*K+k] * B[n*K+k].
// A: MxK row-major f16, B: NxK row-major f16 (both pre-cast/transposed).
// Block tile M=32 x N=64, 4 waves, 2 accumulators per wave.
// Tiles staged to LDS with GLOBAL_LOAD_ASYNC_TO_LDS_B128 when available.
// K must be a multiple of 32 (K=256 here); M multiple of 32.
// ===========================================================================
__global__ __launch_bounds__(128)
void k_gemm_nt_f16(const _Float16* __restrict__ A, const _Float16* __restrict__ B,
                   float* __restrict__ C, int M, int N, int K)
{
    const int tid   = threadIdx.x;
    const int lane  = tid & 31;
    const int wid   = tid >> 5;
    const int mi    = wid & 1;
    const int nj    = wid >> 1;
    const int tileM = blockIdx.x;
    const int tileN = blockIdx.y;

    __shared__ alignas(16) _Float16 As[32][40];
    __shared__ alignas(16) _Float16 Bs[64][40];

    v8f acc0 = {};
    v8f acc1 = {};
    const int hi = lane >> 4;
    const int lo = lane & 15;

    for (int k0 = 0; k0 < K; k0 += 32) {
        // A: 32 rows x 4 x 16B segments = 128 chunks (one per thread)
        {
            int r = tid >> 2, s = tid & 3;
            int gm = tileM * 32 + r; if (gm >= M) gm = M - 1;
            const _Float16* src = &A[(size_t)gm * K + k0 + s * 8];
            _Float16*       dst = &As[r][s * 8];
#if RAFT_ASYNC_LDS
            async_cp16(src, dst);
#else
            *(v8h*)dst = *(const v8h*)src;
#endif
        }
        // B: 64 rows x 4 segments = 256 chunks (two per thread)
        for (int c = tid; c < 256; c += 128) {
            int r = c >> 2, s = c & 3;
            int gn = tileN * 64 + r; if (gn >= N) gn = N - 1;
            const _Float16* src = &B[(size_t)gn * K + k0 + s * 8];
            _Float16*       dst = &Bs[r][s * 8];
#if RAFT_ASYNC_LDS
            async_cp16(src, dst);
#else
            *(v8h*)dst = *(const v8h*)src;
#endif
        }
#if RAFT_ASYNC_LDS
        __builtin_amdgcn_s_wait_asynccnt(0);
#endif
        __syncthreads();

        v16h a  = ld_frag(&As[mi * 16 + lo][0], hi);
        v16h b0 = ld_frag(&Bs[nj * 32 + lo][0], hi);
        v16h b1 = ld_frag(&Bs[nj * 32 + 16 + lo][0], hi);
        acc0 = __builtin_amdgcn_wmma_f32_16x16x32_f16(false, a, false, b0,
                                                      (short)0, acc0, false, false);
        acc1 = __builtin_amdgcn_wmma_f32_16x16x32_f16(false, a, false, b1,
                                                      (short)0, acc1, false, false);
        __syncthreads();
    }

    const int m_base = tileM * 32 + mi * 16;
#pragma unroll
    for (int half = 0; half < 2; ++half) {
        int nn = tileN * 64 + nj * 32 + half * 16 + lo;
        if (nn < N) {
#pragma unroll
            for (int r = 0; r < 8; ++r) {
                int m = m_base + r + 8 * hi;
                if (m < M) C[(size_t)m * N + nn] = half ? acc1[r] : acc0[r];
            }
        }
    }
}

// Cast f32 (K x N, k-major) -> f16 (N x K, n-major) for the GEMM operands.
__global__ void k_cast_tr(const float* __restrict__ src, _Float16* __restrict__ dst,
                          int N, int K)
{
    size_t i = blockIdx.x * (size_t)blockDim.x + threadIdx.x;
    size_t tot = (size_t)N * K;
    if (i >= tot) return;
    int k = i % K;
    int n = i / K;
    dst[i] = (_Float16)src[(size_t)k * N + n];
}

// ===========================================================================
// Instance norm over one (n,c) plane; bnorm with B=1 is identical. In-place.
// ===========================================================================
__global__ __launch_bounds__(256)
void k_inorm(float* __restrict__ x, int HW, int relu)
{
    float* p = x + (size_t)blockIdx.x * HW;
    __shared__ float ssum[256];
    __shared__ float ssq[256];
    float s = 0.f, q = 0.f;
    for (int i = threadIdx.x; i < HW; i += 256) {
        float v = p[i];
        s += v; q += v * v;
    }
    ssum[threadIdx.x] = s; ssq[threadIdx.x] = q;
    __syncthreads();
    for (int o = 128; o > 0; o >>= 1) {
        if (threadIdx.x < o) {
            ssum[threadIdx.x] += ssum[threadIdx.x + o];
            ssq[threadIdx.x]  += ssq[threadIdx.x + o];
        }
        __syncthreads();
    }
    float mean = ssum[0] / (float)HW;
    float var  = ssq[0] / (float)HW - mean * mean;
    float inv  = rsqrtf(var + 1e-5f);
    for (int i = threadIdx.x; i < HW; i += 256) {
        float v = (p[i] - mean) * inv;
        if (relu) v = fmaxf(v, 0.f);
        p[i] = v;
    }
}

// ===========================================================================
// Small VALU kernels
// ===========================================================================
__global__ void k_add_relu(const float* a, const float* b, float* o, size_t n)
{
    size_t i = blockIdx.x * (size_t)blockDim.x + threadIdx.x;
    if (i < n) o[i] = fmaxf(a[i] + b[i], 0.f);
}

__global__ void k_add_inplace(float* a, const float* b, size_t n)
{
    size_t i = blockIdx.x * (size_t)blockDim.x + threadIdx.x;
    if (i < n) a[i] += b[i];
}

__global__ void k_avgpool2(const float* in, float* out, int C, int H, int W)
{
    int OH = H >> 1, OW = W >> 1;
    size_t i = blockIdx.x * (size_t)blockDim.x + threadIdx.x;
    size_t tot = (size_t)C * OH * OW;
    if (i >= tot) return;
    int x = i % OW; size_t t = i / OW;
    int y = t % OH; int c = t / OH;
    const float* p = in + ((size_t)c * H + 2 * y) * W + 2 * x;
    out[i] = 0.25f * (p[0] + p[1] + p[W] + p[W + 1]);
}

// dst (Cd ch) at channel dcoff <- src channels [scoff, scoff+Cc). act:0/1 relu/2 tanh
__global__ void k_slice_copy(float* dst, int Cd, int dcoff,
                             const float* src, int Csrc, int scoff,
                             int Cc, int HW, int act)
{
    size_t i = blockIdx.x * (size_t)blockDim.x + threadIdx.x;
    size_t tot = (size_t)Cc * HW;
    if (i >= tot) return;
    int p = i % HW; int c = i / HW;
    float v = src[((size_t)(scoff + c)) * HW + p];
    if (act == 1) v = fmaxf(v, 0.f);
    else if (act == 2) v = tanhf(v);
    dst[((size_t)(dcoff + c)) * HW + p] = v;
}

// dst channels [dcoff, dcoff+C) = a[i] * b[i]   (r * h for the GRU)
__global__ void k_mul_copy(float* dst, int dcoff, const float* a,
                           const float* b, int C, int HW)
{
    size_t i = blockIdx.x * (size_t)blockDim.x + threadIdx.x;
    size_t tot = (size_t)C * HW;
    if (i >= tot) return;
    int p = i % HW; int c = i / HW;
    dst[((size_t)(dcoff + c)) * HW + p] = a[i] * b[i];
}

__global__ void k_gru_combine(float* h, const float* z, const float* q, size_t n)
{
    size_t i = blockIdx.x * (size_t)blockDim.x + threadIdx.x;
    if (i < n) h[i] = (1.f - z[i]) * h[i] + z[i] * q[i];
}

__global__ void k_init_coords(float* c, int h, int w)
{
    int hw = h * w;
    int i = blockIdx.x * blockDim.x + threadIdx.x;
    if (i >= 2 * hw) return;
    int p = i % hw; int ch = i / hw;
    c[i] = (ch == 0) ? (float)(p % w) : (float)(p / w);
}

__global__ void k_flow(const float* coords, float* flow, int h, int w)
{
    int hw = h * w;
    int i = blockIdx.x * blockDim.x + threadIdx.x;
    if (i >= 2 * hw) return;
    int p = i % hw; int ch = i / hw;
    float g = (ch == 0) ? (float)(p % w) : (float)(p / w);
    flow[i] = coords[i] - g;
}

// ===========================================================================
// Correlation lookup: out[ch][p], ch = level*81 + ix*9 + iy. V: [p][H2][W2].
// ===========================================================================
__global__ void k_corr_lookup(const float* V0, const float* V1,
                              const float* V2, const float* V3,
                              const float* coords, float* out,
                              int h, int w, int rad)
{
    int hw = h * w;
    int D  = 2 * rad + 1;
    size_t i = blockIdx.x * (size_t)blockDim.x + threadIdx.x;
    size_t tot = (size_t)4 * D * D * hw;
    if (i >= tot) return;
    int p  = i % hw;
    int ch = i / hw;
    int level = ch / (D * D);
    int rr = ch % (D * D);
    int ix = rr / D, iy = rr % D;
    const float* V = (level == 0) ? V0 : (level == 1) ? V1 : (level == 2) ? V2 : V3;
    int H2 = h >> level, W2 = w >> level;

    float cx = coords[p];
    float cy = coords[hw + p];
    float sc = 1.f / (float)(1 << level);
    float xs = cx * sc + (float)(ix - rad);
    float ys = cy * sc + (float)(iy - rad);
    float x0 = floorf(xs), y0 = floorf(ys);
    float wx = xs - x0,  wy = ys - y0;
    int xi = (int)x0, yi = (int)y0;
    const float* Vp = V + (size_t)p * H2 * W2;

    float c00 = (xi >= 0 && xi < W2 && yi >= 0 && yi < H2)                 ? Vp[(size_t)yi * W2 + xi]           : 0.f;
    float c01 = (xi + 1 >= 0 && xi + 1 < W2 && yi >= 0 && yi < H2)         ? Vp[(size_t)yi * W2 + xi + 1]       : 0.f;
    float c10 = (xi >= 0 && xi < W2 && yi + 1 >= 0 && yi + 1 < H2)         ? Vp[(size_t)(yi + 1) * W2 + xi]     : 0.f;
    float c11 = (xi + 1 >= 0 && xi + 1 < W2 && yi + 1 >= 0 && yi + 1 < H2) ? Vp[(size_t)(yi + 1) * W2 + xi + 1] : 0.f;

    out[(size_t)ch * hw + p] =
        c00 * (1.f - wy) * (1.f - wx) + c01 * (1.f - wy) * wx +
        c10 * wy * (1.f - wx)         + c11 * wy * wx;
}

// ===========================================================================
// Convex upsample: mask (576,h,w) -> softmax over 9, combine 3x3 of 8*flow.
// ===========================================================================
__global__ void k_upsample(const float* flow, const float* mask,
                           float* out, int h, int w)
{
    int hw = h * w;
    size_t i = blockIdx.x * (size_t)blockDim.x + threadIdx.x;
    size_t tot = (size_t)hw * 64;
    if (i >= tot) return;
    int sub = i & 63; int sx = sub & 7, sy = sub >> 3;
    int p = i >> 6;   int x = p % w,   y = p / w;

    float mv[9];
    float mx = -1e30f;
#pragma unroll
    for (int k = 0; k < 9; ++k) {
        mv[k] = mask[((size_t)(k * 64 + sy * 8 + sx)) * hw + p];
        mx = fmaxf(mx, mv[k]);
    }
    float se = 0.f;
#pragma unroll
    for (int k = 0; k < 9; ++k) { mv[k] = expf(mv[k] - mx); se += mv[k]; }
    float inv = 1.f / se;

    int OW = 8 * w;
#pragma unroll
    for (int c = 0; c < 2; ++c) {
        float acc = 0.f;
#pragma unroll
        for (int ky = 0; ky < 3; ++ky)
#pragma unroll
            for (int kx = 0; kx < 3; ++kx) {
                int yy = y + ky - 1, xx = x + kx - 1;
                float f = (yy >= 0 && yy < h && xx >= 0 && xx < w)
                          ? 8.f * flow[(size_t)c * hw + (size_t)yy * w + xx] : 0.f;
                acc += mv[ky * 3 + kx] * f;
            }
        out[((size_t)c * (8 * h) + (y * 8 + sy)) * (size_t)OW + (x * 8 + sx)] = acc * inv;
    }
}

// ===========================================================================
// Host-side launchers and orchestration
// ===========================================================================
static inline void conv(hipStream_t st, const float* in, const float* w,
                        const float* b, float* out, int N, int Cin, int H,
                        int W, int Cout, int KH, int KW, int s, int pH, int pW,
                        int OH, int OW, int fuse)
{
    dim3 g((unsigned)((OH * OW + 31) / 32), (unsigned)((Cout + 63) / 64), (unsigned)N);
    k_conv_wmma<<<g, 128, 0, st>>>(in, w, b, out, N, Cin, H, W, Cout,
                                   KH, KW, s, pH, pW, OH, OW, fuse);
}

static inline void gemm16(hipStream_t st, const _Float16* A, const _Float16* B,
                          float* C, int M, int N, int K)
{
    dim3 g((unsigned)((M + 31) / 32), (unsigned)((N + 63) / 64), 1);
    k_gemm_nt_f16<<<g, 128, 0, st>>>(A, B, C, M, N, K);
}

static inline void inorm(hipStream_t st, float* x, int planes, int HW, int relu)
{
    k_inorm<<<planes, 256, 0, st>>>(x, HW, relu);
}

static inline unsigned nb(size_t n) { return (unsigned)((n + 255) / 256); }

struct Enc {
    const float *c1w, *c1b, *c2w, *c2b;
    const float *w1[6], *b1[6], *w2[6], *b2[6], *wd[6], *bd[6];
};

static int parse_enc(void* const* d_in, int base, Enc& e)
{
    int i = base;
    e.c1w = (const float*)d_in[i++]; e.c1b = (const float*)d_in[i++];
    const int hasd[6] = {0, 0, 1, 0, 1, 0};
    for (int b = 0; b < 6; ++b) {
        e.w1[b] = (const float*)d_in[i++]; e.b1[b] = (const float*)d_in[i++];
        e.w2[b] = (const float*)d_in[i++]; e.b2[b] = (const float*)d_in[i++];
        if (hasd[b]) { e.wd[b] = (const float*)d_in[i++]; e.bd[b] = (const float*)d_in[i++]; }
        else         { e.wd[b] = nullptr;                  e.bd[b] = nullptr; }
    }
    e.c2w = (const float*)d_in[i++]; e.c2b = (const float*)d_in[i++];
    return i;
}

static void run_encoder(const Enc& e, const float* input, int N, float* out256,
                        float* s0, float* s1, float* s2, hipStream_t st)
{
    conv(st, input, e.c1w, e.c1b, s0, N, 3, 384, 768, 64, 7, 7, 2, 3, 3, 192, 384, 0);
    inorm(st, s0, N * 64, 192 * 384, 1);

    const int ch[7]      = {64, 64, 64, 96, 96, 128, 128};
    const int strides[6] = {1, 1, 2, 1, 2, 1};
    int H = 192, W = 384;
    for (int b = 0; b < 6; ++b) {
        int ci = ch[b], co = ch[b + 1], s = strides[b];
        int OH = H / s, OW = W / s;
        conv(st, s0, e.w1[b], e.b1[b], s1, N, ci, H, W, co, 3, 3, s, 1, 1, OH, OW, 0);
        inorm(st, s1, N * co, OH * OW, 1);
        conv(st, s1, e.w2[b], e.b2[b], s2, N, co, OH, OW, co, 3, 3, 1, 1, 1, OH, OW, 0);
        inorm(st, s2, N * co, OH * OW, 1);
        const float* xres = s0;
        if (e.wd[b]) {
            conv(st, s0, e.wd[b], e.bd[b], s1, N, ci, H, W, co, 1, 1, s, 0, 0, OH, OW, 0);
            inorm(st, s1, N * co, OH * OW, 0);
            xres = s1;
        }
        size_t n = (size_t)N * co * OH * OW;
        k_add_relu<<<nb(n), 256, 0, st>>>(xres, s2, s0, n);
        H = OH; W = OW;
    }
    conv(st, s0, e.c2w, e.c2b, out256, N, 128, 48, 96, 256, 1, 1, 1, 0, 0, 48, 96, 0);
}

extern "C" void kernel_launch(void* const* d_in, const int* in_sizes, int n_in,
                              void* d_out, int out_size, void* d_ws, size_t ws_size,
                              hipStream_t stream)
{
    (void)in_sizes; (void)n_in; (void)out_size; (void)ws_size;
    const float* img1 = (const float*)d_in[0];
    const float* img2 = (const float*)d_in[1];

    Enc fnet, cnet;
    int idx = parse_enc(d_in, 2, fnet);   // -> 34
    idx = parse_enc(d_in, idx, cnet);     // -> 66
#define FP(k) ((const float*)d_in[(k)])
    const float *cc1w = FP(66), *cc1b = FP(67), *cc2w = FP(68), *cc2b = FP(69);
    const float *cf1w = FP(70), *cf1b = FP(71), *cf2w = FP(72), *cf2b = FP(73);
    const float *cmw  = FP(74), *cmb  = FP(75);
    const float *z1w  = FP(76), *z1b  = FP(77), *r1w = FP(78), *r1b = FP(79);
    const float *q1w  = FP(80), *q1b  = FP(81);
    const float *z2w  = FP(82), *z2b  = FP(83), *r2w = FP(84), *r2b = FP(85);
    const float *q2w  = FP(86), *q2b  = FP(87);
    const float *fh1w = FP(88), *fh1b = FP(89), *fh2w = FP(90), *fh2b = FP(91);
    const float *mk1w = FP(92), *mk1b = FP(93), *mk2w = FP(94), *mk2b = FP(95);
#undef FP
    const int iterations = 8;  // fixed (deterministic; graph-capture safe)

    const int h = 48, w = 96, hw = h * w;

    // ---- workspace bump allocator (floats, 16B granular) ----
    float* base = (float*)d_ws;
    size_t off = 0;
    auto al = [&](size_t n) { n = (n + 3) & ~(size_t)3; float* p = base + off; off += n; return p; };

    float* catimg = al((size_t)2 * 3 * 384 * 768);
    float* fmaps  = al((size_t)2 * 256 * hw);
    float* ctx    = al((size_t)256 * hw);
    float* s0     = al((size_t)2 * 64 * 192 * 384);
    float* s1     = al((size_t)2 * 64 * 192 * 384);
    float* s2     = al((size_t)2 * 64 * 192 * 384);
    float* V0     = al((size_t)hw * hw);
    float* V1     = al((size_t)hw * 24 * 48);
    float* V2     = al((size_t)hw * 12 * 24);
    float* V3     = al((size_t)hw * 6 * 12);
    float* f2a    = al((size_t)256 * 24 * 48);
    float* f2b    = al((size_t)256 * 12 * 24);
    float* f2c    = al((size_t)256 * 6 * 12);
    _Float16* A16  = (_Float16*)al((size_t)hw * 256 / 2);
    _Float16* B16a = (_Float16*)al((size_t)hw * 256 / 2);
    _Float16* B16b = (_Float16*)al((size_t)(24 * 48) * 256 / 2);
    _Float16* B16c = (_Float16*)al((size_t)(12 * 24) * 256 / 2);
    _Float16* B16d = (_Float16*)al((size_t)(6 * 12) * 256 / 2);
    float* hst    = al((size_t)128 * hw);
    float* xc     = al((size_t)128 * hw);
    float* coords = al((size_t)2 * hw);
    float* flow   = al((size_t)2 * hw);
    float* corr   = al((size_t)324 * hw);
    float* t256   = al((size_t)256 * hw);
    float* c192   = al((size_t)192 * hw);
    float* f128   = al((size_t)128 * hw);
    float* f64b   = al((size_t)64 * hw);
    float* cat256 = al((size_t)256 * hw);
    float* m126   = al((size_t)126 * hw);
    float* inp    = al((size_t)256 * hw);
    float* hx     = al((size_t)384 * hw);
    float* zb     = al((size_t)128 * hw);
    float* rb     = al((size_t)128 * hw);
    float* qb     = al((size_t)128 * hw);
    float* qin    = al((size_t)384 * hw);
    float* fh     = al((size_t)256 * hw);
    float* dfl    = al((size_t)2 * hw);
    float* mk     = al((size_t)256 * hw);
    float* mask   = al((size_t)576 * hw);

    // ---- 1. feature + context encoders (WMMA implicit-GEMM convs) ----
    size_t imgN = (size_t)3 * 384 * 768;
    (void)hipMemcpyAsync(catimg,        img1, imgN * sizeof(float), hipMemcpyDeviceToDevice, stream);
    (void)hipMemcpyAsync(catimg + imgN, img2, imgN * sizeof(float), hipMemcpyDeviceToDevice, stream);
    run_encoder(fnet, catimg, 2, fmaps, s0, s1, s2, stream);
    run_encoder(cnet, img1,   1, ctx,   s0, s1, s2, stream);
    float* fmap1 = fmaps;
    float* fmap2 = fmaps + (size_t)256 * hw;

    // h = tanh(ctx[:,:128]); xc = relu(ctx[:,128:])
    k_slice_copy<<<nb((size_t)128 * hw), 256, 0, stream>>>(hst, 128, 0, ctx, 256, 0,   128, hw, 2);
    k_slice_copy<<<nb((size_t)128 * hw), 256, 0, stream>>>(xc,  128, 0, ctx, 256, 128, 128, hw, 1);

    // ---- 2. correlation pyramid: f16 WMMA GEMM per level ----
    k_avgpool2<<<nb((size_t)256 * 24 * 48), 256, 0, stream>>>(fmap2, f2a, 256, 48, 96);
    k_avgpool2<<<nb((size_t)256 * 12 * 24), 256, 0, stream>>>(f2a, f2b, 256, 24, 48);
    k_avgpool2<<<nb((size_t)256 * 6 * 12), 256, 0, stream>>>(f2b, f2c, 256, 12, 24);
    k_cast_tr<<<nb((size_t)hw * 256), 256, 0, stream>>>(fmap1, A16, hw, 256);
    k_cast_tr<<<nb((size_t)hw * 256), 256, 0, stream>>>(fmap2, B16a, hw, 256);
    k_cast_tr<<<nb((size_t)(24 * 48) * 256), 256, 0, stream>>>(f2a, B16b, 24 * 48, 256);
    k_cast_tr<<<nb((size_t)(12 * 24) * 256), 256, 0, stream>>>(f2b, B16c, 12 * 24, 256);
    k_cast_tr<<<nb((size_t)(6 * 12) * 256), 256, 0, stream>>>(f2c, B16d, 6 * 12, 256);
    gemm16(stream, A16, B16a, V0, hw, hw, 256);
    gemm16(stream, A16, B16b, V1, hw, 24 * 48, 256);
    gemm16(stream, A16, B16c, V2, hw, 12 * 24, 256);
    gemm16(stream, A16, B16d, V3, hw, 6 * 12, 256);

    k_init_coords<<<nb((size_t)2 * hw), 256, 0, stream>>>(coords, h, w);

    float* outp = (float*)d_out;
    const size_t iter_stride = (size_t)2 * 384 * 768;

    for (int it = 0; it < iterations; ++it) {
        k_corr_lookup<<<nb((size_t)324 * hw), 256, 0, stream>>>(V0, V1, V2, V3, coords, corr, h, w, 4);
        k_flow<<<nb((size_t)2 * hw), 256, 0, stream>>>(coords, flow, h, w);

        // motion encoder
        conv(stream, corr, cc1w, cc1b, t256, 1, 324, h, w, 256, 1, 1, 1, 0, 0, h, w, 1);
        conv(stream, t256, cc2w, cc2b, c192, 1, 256, h, w, 192, 3, 3, 1, 1, 1, h, w, 1);
        conv(stream, flow, cf1w, cf1b, f128, 1, 2,   h, w, 128, 7, 7, 1, 3, 3, h, w, 1);
        conv(stream, f128, cf2w, cf2b, f64b, 1, 128, h, w, 64,  3, 3, 1, 1, 1, h, w, 1);
        k_slice_copy<<<nb((size_t)192 * hw), 256, 0, stream>>>(cat256, 256, 0,   c192, 192, 0, 192, hw, 0);
        k_slice_copy<<<nb((size_t)64  * hw), 256, 0, stream>>>(cat256, 256, 192, f64b, 64,  0, 64,  hw, 0);
        conv(stream, cat256, cmw, cmb, m126, 1, 256, h, w, 126, 3, 3, 1, 1, 1, h, w, 1);

        // inp = concat(xc, m126, flow)
        k_slice_copy<<<nb((size_t)128 * hw), 256, 0, stream>>>(inp, 256, 0,   xc,   128, 0, 128, hw, 0);
        k_slice_copy<<<nb((size_t)126 * hw), 256, 0, stream>>>(inp, 256, 128, m126, 126, 0, 126, hw, 0);
        k_slice_copy<<<nb((size_t)2   * hw), 256, 0, stream>>>(inp, 256, 254, flow, 2,   0, 2,   hw, 0);

        // --- GRU, horizontal (1x5, pad W=2) ---
        k_slice_copy<<<nb((size_t)128 * hw), 256, 0, stream>>>(hx, 384, 0,   hst, 128, 0, 128, hw, 0);
        k_slice_copy<<<nb((size_t)256 * hw), 256, 0, stream>>>(hx, 384, 128, inp, 256, 0, 256, hw, 0);
        conv(stream, hx, z1w, z1b, zb, 1, 384, h, w, 128, 1, 5, 1, 0, 2, h, w, 2);
        conv(stream, hx, r1w, r1b, rb, 1, 384, h, w, 128, 1, 5, 1, 0, 2, h, w, 2);
        k_mul_copy<<<nb((size_t)128 * hw), 256, 0, stream>>>(qin, 0, rb, hst, 128, hw);
        k_slice_copy<<<nb((size_t)256 * hw), 256, 0, stream>>>(qin, 384, 128, inp, 256, 0, 256, hw, 0);
        conv(stream, qin, q1w, q1b, qb, 1, 384, h, w, 128, 1, 5, 1, 0, 2, h, w, 3);
        k_gru_combine<<<nb((size_t)128 * hw), 256, 0, stream>>>(hst, zb, qb, (size_t)128 * hw);

        // --- GRU, vertical (5x1, pad H=2) ---
        k_slice_copy<<<nb((size_t)128 * hw), 256, 0, stream>>>(hx, 384, 0,   hst, 128, 0, 128, hw, 0);
        k_slice_copy<<<nb((size_t)256 * hw), 256, 0, stream>>>(hx, 384, 128, inp, 256, 0, 256, hw, 0);
        conv(stream, hx, z2w, z2b, zb, 1, 384, h, w, 128, 5, 1, 1, 2, 0, h, w, 2);
        conv(stream, hx, r2w, r2b, rb, 1, 384, h, w, 128, 5, 1, 1, 2, 0, h, w, 2);
        k_mul_copy<<<nb((size_t)128 * hw), 256, 0, stream>>>(qin, 0, rb, hst, 128, hw);
        k_slice_copy<<<nb((size_t)256 * hw), 256, 0, stream>>>(qin, 384, 128, inp, 256, 0, 256, hw, 0);
        conv(stream, qin, q2w, q2b, qb, 1, 384, h, w, 128, 5, 1, 1, 2, 0, h, w, 3);
        k_gru_combine<<<nb((size_t)128 * hw), 256, 0, stream>>>(hst, zb, qb, (size_t)128 * hw);

        // --- flow head + mask head ---
        conv(stream, hst, fh1w, fh1b, fh,  1, 128, h, w, 256, 3, 3, 1, 1, 1, h, w, 1);
        conv(stream, fh,  fh2w, fh2b, dfl, 1, 256, h, w, 2,   3, 3, 1, 1, 1, h, w, 0);
        conv(stream, hst, mk1w, mk1b, mk,  1, 128, h, w, 256, 3, 3, 1, 1, 1, h, w, 1);
        conv(stream, mk,  mk2w, mk2b, mask, 1, 256, h, w, 576, 1, 1, 1, 0, 0, h, w, 4);

        // coords1 += d ; upsample(coords1 - coords0, mask)
        k_add_inplace<<<nb((size_t)2 * hw), 256, 0, stream>>>(coords, dfl, (size_t)2 * hw);
        k_flow<<<nb((size_t)2 * hw), 256, 0, stream>>>(coords, flow, h, w);
        k_upsample<<<nb((size_t)hw * 64), 256, 0, stream>>>(flow, mask, outp + (size_t)it * iter_stride, h, w);
    }
}
